// IPRFormer_43078521979438
// MI455X (gfx1250) — matrix operand order repaired and checked by reading, not compiled, not run on hardware
//
#include <hip/hip_runtime.h>
#include <hip/hip_bf16.h>

// Problem constants (from reference): B=1024, M=100, N=40000, D=2048, C=5000
#define B_ 1024
#define M_ 100
#define N_ 40000
#define D_ 2048
#define C_ 5000

typedef __attribute__((ext_vector_type(16))) __bf16 v16bf;
typedef __attribute__((ext_vector_type(8)))  float  v8f;
typedef __attribute__((ext_vector_type(4)))  __bf16 bf4;

union BFVec { v16bf v; uint4 q[2]; };

// Workspace layout (bytes, 256-aligned):
//   vidx : B*M int32                =   409,600
//   h_hi : B*D bf16                 = 4,194,304
//   h_lo : B*D bf16                 = 4,194,304
//   w2hi : C*D bf16                 = 20,480,000
//   w2lo : C*D bf16                 = 20,480,000
#define OFF_IDX   0u
#define OFF_HHI   409600u
#define OFF_HLO   4603904u
#define OFF_W2HI  8798208u
#define OFF_W2LO  29278208u

// ---------------------------------------------------------------------------
// Kernel 0: dedupe indices + drop column N (reference scatters 1.0 via .set,
// so duplicate indices collapse; index N lands in the dropped column).
// vidx[b][m] = x[b][m] if it's the first occurrence in row b and != N, else -1
// ---------------------------------------------------------------------------
__global__ void k0_mask(const int* __restrict__ x, int* __restrict__ vidx) {
  __shared__ int row[M_];
  const int b = blockIdx.x;
  const int t = threadIdx.x;
  if (t < M_) row[t] = x[b * M_ + t];
  __syncthreads();
  if (t < M_) {
    const int v = row[t];
    bool keep = (v != N_);
    for (int j = 0; j < t; ++j) keep = keep && (row[j] != v);
    vidx[b * M_ + t] = keep ? v : -1;
  }
}

// ---------------------------------------------------------------------------
// Kernel 1: split W2 (fp32) into bf16 hi/lo planes for the hi/lo WMMA trick.
// ---------------------------------------------------------------------------
__global__ void k1_split_w2(const float* __restrict__ w2,
                            __bf16* __restrict__ hi, __bf16* __restrict__ lo) {
  const int i = (blockIdx.x * blockDim.x + threadIdx.x) * 4;
  const float4 f = *(const float4*)(w2 + i);
  float v[4] = { f.x, f.y, f.z, f.w };
  bf4 h4, l4;
#pragma unroll
  for (int j = 0; j < 4; ++j) {
    __bf16 h = (__bf16)v[j];
    h4[j] = h;
    l4[j] = (__bf16)(v[j] - (float)h);
  }
  *(bf4*)(hi + i) = h4;
  *(bf4*)(lo + i) = l4;
}

// ---------------------------------------------------------------------------
// Kernel 2: layer 1 gather-sum. One workgroup per output feature d.
// Stages the full 160KB row W1[d,:] in LDS (CDNA5: 320KB/WGP) plus the
// dedup'd index lists for a 256-row batch chunk (100KB). Each thread owns one
// batch row per chunk: h[b][d] = b1[d] + sum(row[idx]). Result stored as bf16
// hi/lo planes, ready for the WMMA GEMM.
// ---------------------------------------------------------------------------
__global__ void __launch_bounds__(256, 1) k2_layer1(
    const float* __restrict__ W1, const float* __restrict__ b1,
    const int* __restrict__ vidx,
    __bf16* __restrict__ hhi, __bf16* __restrict__ hlo) {
  __shared__ float row[N_];          // 160,000 B
  __shared__ int   lidx[256 * M_];   // 102,400 B
  const int d = blockIdx.x;
  const int t = threadIdx.x;

  const float4* w4 = (const float4*)(W1 + (size_t)d * N_);
  for (int i = t; i < N_ / 4; i += 256) ((float4*)row)[i] = w4[i];
  const float bias = b1[d];

  for (int chunk = 0; chunk < B_ / 256; ++chunk) {
    __syncthreads();  // row ready (chunk 0) / previous chunk's reads done
    const int* src = vidx + (size_t)chunk * 256 * M_;
    for (int i = t; i < 256 * M_; i += 256) lidx[i] = src[i];
    __syncthreads();

    float acc = bias;
    const int* my = lidx + t * M_;
#pragma unroll 4
    for (int j = 0; j < M_; ++j) {
      const int id = my[j];
      if (id >= 0) acc += row[id];
    }
    const int b = chunk * 256 + t;
    const __bf16 h = (__bf16)acc;
    hhi[(size_t)b * D_ + d] = h;
    hlo[(size_t)b * D_ + d] = (__bf16)(acc - (float)h);
  }
}

// ---------------------------------------------------------------------------
// Kernel 3: out = h @ W2^T + b2 via v_wmma_f32_16x16x32_bf16 with hi/lo
// splitting (hi*hi + hi*lo + lo*hi) for fp32-class accuracy.
// 2x2 register blocking: each wave owns a 32(b) x 32(c) output tile with 4
// independent WMMA accumulator chains -> 16 b128 loads per 12 WMMAs (1.33
// loads/WMMA vs 4 unblocked), halving L2 operand traffic and giving the
// matrix pipe independent chains to overlap.
//   A k order used here (not the ISA table order): lane half hA: elems 0-15 <-> K=hA*16+e
//   B k order used here (same as A): lane half hB: elems 0-15 <-> K=hB*16+e
//   C/D: col n = lane&15, row m = vgpr + 8*(lane>>4)
// ---------------------------------------------------------------------------
__global__ void __launch_bounds__(256) k3_gemm(
    const __bf16* __restrict__ hhi, const __bf16* __restrict__ hlo,
    const __bf16* __restrict__ whi, const __bf16* __restrict__ wlo,
    const float* __restrict__ b2, float* __restrict__ out) {
  const int lane = threadIdx.x;                  // 0..31
  const int tile = blockIdx.x * 8 + threadIdx.y; // 0..5023
  const int tb = tile & 31;                      // 32 b-tiles of 32
  const int tc = tile >> 5;                      // 157 c-tiles of 32
  const int half = lane >> 4;
  const int l16  = lane & 15;

  const int bR0 = tb * 32 + l16;                 // A rows (always < B_)
  const int bR1 = bR0 + 16;
  const int cC0 = tc * 32 + l16;                 // B cols
  const int cC1 = cC0 + 16;
  const int c0c = (cC0 < C_) ? cC0 : (C_ - 1);   // clamp edge loads
  const int c1c = (cC1 < C_) ? cC1 : (C_ - 1);

  const __bf16* aHi0 = hhi + (size_t)bR0 * D_;
  const __bf16* aLo0 = hlo + (size_t)bR0 * D_;
  const __bf16* aHi1 = hhi + (size_t)bR1 * D_;
  const __bf16* aLo1 = hlo + (size_t)bR1 * D_;
  const __bf16* bHi0 = whi + (size_t)c0c * D_;
  const __bf16* bLo0 = wlo + (size_t)c0c * D_;
  const __bf16* bHi1 = whi + (size_t)c1c * D_;
  const __bf16* bLo1 = wlo + (size_t)c1c * D_;

  const int aOff0 = half * 16;
  const int aOff1 = half * 16 + 8;
  const int bOff  = half * 16;

  v8f acc00 = {}, acc01 = {}, acc10 = {}, acc11 = {};

  for (int kc = 0; kc < D_; kc += 32) {
    BFVec ah0, al0, ah1, al1, bh0, bl0, bh1, bl1;
    ah0.q[0] = *(const uint4*)(aHi0 + kc + aOff0);
    ah0.q[1] = *(const uint4*)(aHi0 + kc + aOff1);
    al0.q[0] = *(const uint4*)(aLo0 + kc + aOff0);
    al0.q[1] = *(const uint4*)(aLo0 + kc + aOff1);
    ah1.q[0] = *(const uint4*)(aHi1 + kc + aOff0);
    ah1.q[1] = *(const uint4*)(aHi1 + kc + aOff1);
    al1.q[0] = *(const uint4*)(aLo1 + kc + aOff0);
    al1.q[1] = *(const uint4*)(aLo1 + kc + aOff1);
    bh0.q[0] = *(const uint4*)(bHi0 + kc + bOff);
    bh0.q[1] = *(const uint4*)(bHi0 + kc + bOff + 8);
    bl0.q[0] = *(const uint4*)(bLo0 + kc + bOff);
    bl0.q[1] = *(const uint4*)(bLo0 + kc + bOff + 8);
    bh1.q[0] = *(const uint4*)(bHi1 + kc + bOff);
    bh1.q[1] = *(const uint4*)(bHi1 + kc + bOff + 8);
    bl1.q[0] = *(const uint4*)(bLo1 + kc + bOff);
    bl1.q[1] = *(const uint4*)(bLo1 + kc + bOff + 8);

    // hi*hi terms (4 independent chains interleaved)
    acc00 = __builtin_amdgcn_wmma_f32_16x16x32_bf16(false, ah0.v, false, bh0.v,
                                                    (short)0, acc00, false, false);
    acc10 = __builtin_amdgcn_wmma_f32_16x16x32_bf16(false, ah1.v, false, bh0.v,
                                                    (short)0, acc10, false, false);
    acc01 = __builtin_amdgcn_wmma_f32_16x16x32_bf16(false, ah0.v, false, bh1.v,
                                                    (short)0, acc01, false, false);
    acc11 = __builtin_amdgcn_wmma_f32_16x16x32_bf16(false, ah1.v, false, bh1.v,
                                                    (short)0, acc11, false, false);
    // hi*lo terms
    acc00 = __builtin_amdgcn_wmma_f32_16x16x32_bf16(false, ah0.v, false, bl0.v,
                                                    (short)0, acc00, false, false);
    acc10 = __builtin_amdgcn_wmma_f32_16x16x32_bf16(false, ah1.v, false, bl0.v,
                                                    (short)0, acc10, false, false);
    acc01 = __builtin_amdgcn_wmma_f32_16x16x32_bf16(false, ah0.v, false, bl1.v,
                                                    (short)0, acc01, false, false);
    acc11 = __builtin_amdgcn_wmma_f32_16x16x32_bf16(false, ah1.v, false, bl1.v,
                                                    (short)0, acc11, false, false);
    // lo*hi terms
    acc00 = __builtin_amdgcn_wmma_f32_16x16x32_bf16(false, al0.v, false, bh0.v,
                                                    (short)0, acc00, false, false);
    acc10 = __builtin_amdgcn_wmma_f32_16x16x32_bf16(false, al1.v, false, bh0.v,
                                                    (short)0, acc10, false, false);
    acc01 = __builtin_amdgcn_wmma_f32_16x16x32_bf16(false, al0.v, false, bh1.v,
                                                    (short)0, acc01, false, false);
    acc11 = __builtin_amdgcn_wmma_f32_16x16x32_bf16(false, al1.v, false, bh1.v,
                                                    (short)0, acc11, false, false);
  }

  // Store 2x2 subtiles: col n = lane&15, row m = v + 8*half within subtile.
#pragma unroll
  for (int j = 0; j < 2; ++j) {
    const int c = tc * 32 + j * 16 + l16;
    if (c < C_) {
      const float bias = b2[c];
      const v8f* accs[2] = { j ? &acc01 : &acc00, j ? &acc11 : &acc10 };
#pragma unroll
      for (int i = 0; i < 2; ++i) {
#pragma unroll
        for (int v = 0; v < 8; ++v) {
          const int bRow = tb * 32 + i * 16 + v + 8 * half;
          out[(size_t)bRow * C_ + c] = (*accs[i])[v] + bias;
        }
      }
    }
  }
}

// ---------------------------------------------------------------------------
extern "C" void kernel_launch(void* const* d_in, const int* in_sizes, int n_in,
                              void* d_out, int out_size, void* d_ws, size_t ws_size,
                              hipStream_t stream) {
  (void)in_sizes; (void)n_in; (void)out_size; (void)ws_size;
  const int*   x  = (const int*)  d_in[0];
  const float* W1 = (const float*)d_in[1];
  const float* b1 = (const float*)d_in[2];
  const float* W2 = (const float*)d_in[3];
  const float* b2 = (const float*)d_in[4];
  float* out = (float*)d_out;

  char* ws = (char*)d_ws;
  int*    vidx = (int*)   (ws + OFF_IDX);
  __bf16* hhi  = (__bf16*)(ws + OFF_HHI);
  __bf16* hlo  = (__bf16*)(ws + OFF_HLO);
  __bf16* w2hi = (__bf16*)(ws + OFF_W2HI);
  __bf16* w2lo = (__bf16*)(ws + OFF_W2LO);

  k0_mask<<<B_, 128, 0, stream>>>(x, vidx);
  k1_split_w2<<<(C_ * D_) / (256 * 4), 256, 0, stream>>>(W2, w2hi, w2lo);
  k2_layer1<<<D_, 256, 0, stream>>>(W1, b1, vidx, hhi, hlo);
  // 32 b-tiles * 157 c-tiles = 5024 wave-tiles, 8 waves per 256-thread block
  k3_gemm<<<628, dim3(32, 8), 0, stream>>>(hhi, hlo, w2hi, w2lo, b2, out);
}
